// PrismV2_83562883711463
// MI455X (gfx1250) — matrix operand order repaired and checked by reading, not compiled, not run on hardware
//
#include <hip/hip_runtime.h>
#include <hip/hip_bf16.h>
#include <math.h>

#define CDIM 128
#define NHEADS 4
#define HD 32
#define NEXP 16
#define EHID 256
#define TOK_TOTAL 32768
#define NWIN 512
#define LN_EPS 1e-5f

typedef __attribute__((ext_vector_type(16))) __bf16        v16bf;
typedef __attribute__((ext_vector_type(8)))  float         v8f;
typedef __attribute__((ext_vector_type(4)))  float         v4f;
typedef __attribute__((ext_vector_type(4)))  unsigned int  v4u;
typedef __attribute__((ext_vector_type(2)))  unsigned int  v2u;

union FragBF {
  v16bf v;
  v4u q[2];
  unsigned short u[16];
};

__device__ __forceinline__ unsigned short f2bf(float f) {
  unsigned int u = __float_as_uint(f);
  unsigned int r = (u + 0x7FFFu + ((u >> 16) & 1u)) >> 16;  // RNE
  return (unsigned short)r;
}

__device__ __forceinline__ v8f wmma_bf16(const FragBF& a, const FragBF& b, v8f c) {
  return __builtin_amdgcn_wmma_f32_16x16x32_bf16(false, a.v, false, b.v,
                                                 (short)0, c, false, false);
}

// ---- A fragment (16x32, MxK) from row-major [M][K] bf16 source ----
// lane group g: element i<8 -> K=k0+g*8+i ; i>=8 -> K=k0+16+g*8+(i-8)
__device__ __forceinline__ void ldA_bf16(FragBF& fr, const unsigned short* src,
                                         int m0, int k0, int ld, int lane) {
  int lo = lane & 15, g = lane >> 4;
  const v4u* p = (const v4u*)(src + (m0 + lo) * ld + k0 + g * 8);
  fr.q[0] = p[0];   // K = k0+g*8    .. +7
  fr.q[1] = p[2];   // K = k0+16+g*8 .. +7
}

// ---- B fragment (32x16, KxN) where B[k][n] = W[n][k], W row-major [N][K] ----
__device__ __forceinline__ void ldB_nk_bf16(FragBF& fr, const unsigned short* W,
                                            int n0, int k0, int ld, int lane) {
  int lo = lane & 15, g = lane >> 4;
  const v4u* p = (const v4u*)(W + (size_t)(n0 + lo) * ld + k0 + g * 8);
  fr.q[0] = p[0];
  fr.q[1] = p[2];
}

// ============================================================================
// Weight pre-conversion: fp32 -> bf16 (run once per call; memory-bound)
// ============================================================================
__global__ void __launch_bounds__(256)
cvt_kernel(const float* __restrict__ s, unsigned short* __restrict__ d, int n) {
  int i = (blockIdx.x * 256 + threadIdx.x) * 4;
  if (i >= n) return;
  v4f v = *(const v4f*)(s + i);
  v2u o;
  o.x = (unsigned int)f2bf(v.x) | ((unsigned int)f2bf(v.y) << 16);
  o.y = (unsigned int)f2bf(v.z) | ((unsigned int)f2bf(v.w) << 16);
  *(v2u*)(d + i) = o;
}

// ============================================================================
// Kernel A: LN1 + windowed attention + residual.  One block per 8x8 window.
// Dynamic LDS (160 KB):
//   [0     ) xln  bf16 64x128   (16 KB)  -- reused as attn-out
//   [16384 ) qk_s bf16 64x256   (32 KB)
//   [49152 ) vT   bf16 128x64   (16 KB)  -- v transposed [chan][tok]
//   [65536 ) sc   f32  4x64x64  (64 KB)
//   [131072) pb   bf16 4x64x64  (32 KB)  -- softmax probs
// ============================================================================
__global__ void __launch_bounds__(256)
attn_kernel(const float* __restrict__ x, const float* __restrict__ g1,
            const float* __restrict__ bt1,
            const unsigned short* __restrict__ wq_bf,   // 384x128 bf16
            const float* __restrict__ b_qkv,
            const unsigned short* __restrict__ wp_bf,   // 128x128 bf16
            const float* __restrict__ b_proj, float* __restrict__ x1) {
  extern __shared__ unsigned char smem[];
  unsigned short* xln  = (unsigned short*)smem;
  unsigned short* qk_s = (unsigned short*)(smem + 16384);
  unsigned short* vT   = (unsigned short*)(smem + 49152);
  float*          sc   = (float*)(smem + 65536);
  unsigned short* pb   = (unsigned short*)(smem + 131072);

  const int lane = threadIdx.x & 31;
  const int wave = threadIdx.x >> 5;
  const int w    = blockIdx.x;
  const int bb   = w >> 8;
  const int rowbase = ((w >> 4) & 15) * 8;
  const int colbase = (w & 15) * 8;

  __builtin_prefetch(wq_bf + threadIdx.x * 192, 0, 0);
  __builtin_prefetch(wp_bf + threadIdx.x * 64, 0, 0);

  // ---- LN1: wave per token ----
  for (int it = 0; it < 8; ++it) {
    int t = it * 8 + wave;
    int gtok = bb * 16384 + (rowbase + (t >> 3)) * 128 + colbase + (t & 7);
    const float* xp = x + (size_t)gtok * CDIM;
    float v0 = xp[lane], v1 = xp[lane + 32], v2 = xp[lane + 64], v3 = xp[lane + 96];
    float s = v0 + v1 + v2 + v3;
    float q = v0 * v0 + v1 * v1 + v2 * v2 + v3 * v3;
#pragma unroll
    for (int m = 16; m >= 1; m >>= 1) {
      s += __shfl_xor(s, m, 32);
      q += __shfl_xor(q, m, 32);
    }
    float mean = s * (1.f / 128.f);
    float var  = q * (1.f / 128.f) - mean * mean;
    float rstd = rsqrtf(var + LN_EPS);
    float vv[4] = {v0, v1, v2, v3};
#pragma unroll
    for (int j = 0; j < 4; ++j) {
      int c = lane + 32 * j;
      xln[t * CDIM + c] = f2bf((vv[j] - mean) * rstd * g1[c] + bt1[c]);
    }
  }
  __syncthreads();

  // ---- QKV: (64x128)@(128x384), 96 tiles; q,k -> qk_s; v -> vT ----
  for (int tile = wave; tile < 96; tile += 8) {
    int mt = tile / 24, nt = tile % 24;
    v8f acc = {0.f, 0.f, 0.f, 0.f, 0.f, 0.f, 0.f, 0.f};
    FragBF a, b;
#pragma unroll
    for (int kt = 0; kt < 4; ++kt) {
      ldA_bf16(a, xln, mt * 16, kt * 32, 128, lane);
      ldB_nk_bf16(b, wq_bf, nt * 16, kt * 32, 128, lane);
      acc = wmma_bf16(a, b, acc);
    }
    int n = nt * 16 + (lane & 15);
    float bias = b_qkv[n];
    int hi = lane >> 4;
    if (nt < 16) {  // q,k row-major [tok][chan]
#pragma unroll
      for (int r = 0; r < 8; ++r)
        qk_s[(mt * 16 + r + 8 * hi) * 256 + n] = f2bf(acc[r] + bias);
    } else {        // v transposed [chan][tok]
      int cv = n - 256;
#pragma unroll
      for (int r = 0; r < 8; ++r)
        vT[cv * 64 + mt * 16 + r + 8 * hi] = f2bf(acc[r] + bias);
    }
  }
  __syncthreads();

  // ---- scores = q k^T / sqrt(hd) ----
  const float scale = 0.17677669529663687f;
  for (int tile = wave; tile < 64; tile += 8) {
    int h = tile >> 4, rem = tile & 15, mt = rem >> 2, nt = rem & 3;
    v8f acc = {0.f, 0.f, 0.f, 0.f, 0.f, 0.f, 0.f, 0.f};
    FragBF a, b;
    ldA_bf16(a, qk_s, mt * 16, h * 32, 256, lane);            // q
    ldB_nk_bf16(b, qk_s, nt * 16, 128 + h * 32, 256, lane);   // k^T
    acc = wmma_bf16(a, b, acc);
    int n = nt * 16 + (lane & 15);
    int hi = lane >> 4;
    float* sp = sc + h * 4096;
#pragma unroll
    for (int r = 0; r < 8; ++r)
      sp[(mt * 16 + r + 8 * hi) * 64 + n] = acc[r] * scale;
  }
  __syncthreads();

  // ---- softmax: one thread per (head,row); emit bf16 probs ----
  {
    int h = threadIdx.x >> 6, m = threadIdx.x & 63;
    float* row = sc + h * 4096 + m * 64;
    unsigned short* prow = pb + h * 4096 + m * 64;
    float mx = row[0];
    for (int i = 1; i < 64; ++i) mx = fmaxf(mx, row[i]);
    float se = 0.f;
    for (int i = 0; i < 64; ++i) { float e = expf(row[i] - mx); row[i] = e; se += e; }
    float inv = 1.f / se;
    for (int i = 0; i < 64; ++i) prow[i] = f2bf(row[i] * inv);
  }
  __syncthreads();

  // ---- AV: per head (64x64)@(64x32); write attn-out over xln ----
  for (int tile = wave; tile < 32; tile += 8) {
    int h = tile >> 3, rem = tile & 7, mt = rem >> 1, nt = rem & 1;
    v8f acc = {0.f, 0.f, 0.f, 0.f, 0.f, 0.f, 0.f, 0.f};
    FragBF a, b;
#pragma unroll
    for (int kt = 0; kt < 2; ++kt) {
      ldA_bf16(a, pb + h * 4096, mt * 16, kt * 32, 64, lane);
      ldB_nk_bf16(b, vT, h * 32 + nt * 16, kt * 32, 64, lane);
      acc = wmma_bf16(a, b, acc);
    }
    int n = h * 32 + nt * 16 + (lane & 15);
    int hi = lane >> 4;
#pragma unroll
    for (int r = 0; r < 8; ++r)
      xln[(mt * 16 + r + 8 * hi) * 128 + n] = f2bf(acc[r]);
  }
  __syncthreads();

  // ---- proj + bias + residual ----
  for (int tile = wave; tile < 32; tile += 8) {
    int mt = tile >> 3, nt = tile & 7;
    v8f acc = {0.f, 0.f, 0.f, 0.f, 0.f, 0.f, 0.f, 0.f};
    FragBF a, b;
#pragma unroll
    for (int kt = 0; kt < 4; ++kt) {
      ldA_bf16(a, xln, mt * 16, kt * 32, 128, lane);
      ldB_nk_bf16(b, wp_bf, nt * 16, kt * 32, 128, lane);
      acc = wmma_bf16(a, b, acc);
    }
    int n = nt * 16 + (lane & 15);
    float bias = b_proj[n];
    int hi = lane >> 4;
#pragma unroll
    for (int r = 0; r < 8; ++r) {
      int m = mt * 16 + r + 8 * hi;
      int gtok = bb * 16384 + (rowbase + (m >> 3)) * 128 + colbase + (m & 7);
      size_t gi = (size_t)gtok * CDIM + n;
      x1[gi] = acc[r] + bias + x[gi];
    }
  }
}

// ============================================================================
// Kernel B: LN2 + WMMA gate + top-1 routing.  Block = 64 tokens.
// ============================================================================
__global__ void __launch_bounds__(256)
route_kernel(const float* __restrict__ x1, const float* __restrict__ g2,
             const float* __restrict__ bt2,
             const unsigned short* __restrict__ gw_bf,   // 16x128 bf16
             unsigned short* __restrict__ xln2, float* __restrict__ wgt,
             int* __restrict__ counts, int* __restrict__ buckets) {
  __shared__ unsigned short xb[64 * 128];   // normalized tokens, bf16
  __shared__ float lg[64 * NEXP];           // gate logits
  int lane = threadIdx.x & 31, wave = threadIdx.x >> 5;
  int tok0 = blockIdx.x * 64;

  // ---- LN2: wave per token ----
  for (int it = 0; it < 8; ++it) {
    int t = it * 8 + wave;
    const float* xp = x1 + (size_t)(tok0 + t) * CDIM;
    float v0 = xp[lane], v1 = xp[lane + 32], v2 = xp[lane + 64], v3 = xp[lane + 96];
    float s = v0 + v1 + v2 + v3;
    float q = v0 * v0 + v1 * v1 + v2 * v2 + v3 * v3;
#pragma unroll
    for (int m = 16; m >= 1; m >>= 1) {
      s += __shfl_xor(s, m, 32);
      q += __shfl_xor(q, m, 32);
    }
    float mean = s * (1.f / 128.f);
    float var  = q * (1.f / 128.f) - mean * mean;
    float rstd = rsqrtf(var + LN_EPS);
    float vv[4] = {v0, v1, v2, v3};
#pragma unroll
    for (int j = 0; j < 4; ++j) {
      int c = lane + 32 * j;
      xb[t * CDIM + c] = f2bf((vv[j] - mean) * rstd * g2[c] + bt2[c]);
    }
  }
  __syncthreads();

  // ---- coalesced b128 copy of normalized tokens to global ----
  {
    const v4u* s = (const v4u*)xb;
    v4u* d = (v4u*)(xln2 + (size_t)tok0 * CDIM);
#pragma unroll
    for (int i = 0; i < 4; ++i)
      d[threadIdx.x + 256 * i] = s[threadIdx.x + 256 * i];
  }

  // ---- gate: (64x128)@(128x16) with WMMA, 4 M-tiles ----
  if (wave < 4) {
    int mt = wave;
    v8f acc = {0.f, 0.f, 0.f, 0.f, 0.f, 0.f, 0.f, 0.f};
    FragBF a, b;
#pragma unroll
    for (int kt = 0; kt < 4; ++kt) {
      ldA_bf16(a, xb, mt * 16, kt * 32, 128, lane);
      ldB_nk_bf16(b, gw_bf, 0, kt * 32, 128, lane);
      acc = wmma_bf16(a, b, acc);
    }
    int n = lane & 15, hi = lane >> 4;
#pragma unroll
    for (int r = 0; r < 8; ++r)
      lg[(mt * 16 + r + 8 * hi) * NEXP + n] = acc[r];
  }
  __syncthreads();

  // ---- top-1 per token ----
  if (threadIdx.x < 64) {
    const float* row = lg + threadIdx.x * NEXP;
    float mx = row[0];
    int am = 0;
#pragma unroll
    for (int e = 1; e < NEXP; ++e)
      if (row[e] > mx) { mx = row[e]; am = e; }
    float se = 0.f;
#pragma unroll
    for (int e = 0; e < NEXP; ++e) se += expf(row[e] - mx);
    float p1 = 1.f / se;
    float wg = p1 / (p1 + 1e-8f);
    int tok = tok0 + (int)threadIdx.x;
    int pos = atomicAdd(counts + am, 1);
    buckets[am * TOK_TOTAL + pos] = tok;
    wgt[tok] = wg;
  }
}

// ============================================================================
// Kernel C: gathered top-1 expert FFN (bf16 weights), scaled + residual.
// Token gather uses GLOBAL_LOAD_ASYNC_TO_LDS_B128 (ASYNCcnt path).
// ============================================================================
__global__ void __launch_bounds__(256)
moe_kernel(const unsigned short* __restrict__ xln2, const float* __restrict__ x1,
           const float* __restrict__ wgt, const int* __restrict__ counts,
           const int* __restrict__ buckets,
           const unsigned short* __restrict__ w1_bf,   // NEXP x 256 x 128 bf16
           const float* __restrict__ b1e,
           const unsigned short* __restrict__ w2_bf,   // NEXP x 128 x 256 bf16
           const float* __restrict__ b2e, float* __restrict__ out) {
  __shared__ int tok_s[16];
  __shared__ float wgt_s[16];
  __shared__ unsigned short a_s[16 * 128];
  __shared__ unsigned short h_s[16 * 256];

  int e = blockIdx.x;
  int cnt = counts[e];
  int base = blockIdx.y * 16;
  if (base >= cnt) return;

  int lane = threadIdx.x & 31, wave = threadIdx.x >> 5;
  if (threadIdx.x < 16) {
    int i = base + (int)threadIdx.x;
    int t = (i < cnt) ? buckets[e * TOK_TOTAL + i] : -1;
    tok_s[threadIdx.x] = t;
    wgt_s[threadIdx.x] = (t >= 0) ? wgt[t] : 0.f;
  }
  __syncthreads();

  // ---- async gather of 16 token rows (256B each) into LDS ----
  {
    int row = threadIdx.x >> 4, cg = threadIdx.x & 15;
    int t = tok_s[row];
    unsigned short* dst = a_s + row * 128 + cg * 8;           // 16B per lane
    unsigned int ldsoff = (unsigned int)(unsigned long long)(void*)dst;
    if (t >= 0) {
      const unsigned short* src = xln2 + (size_t)t * CDIM + cg * 8;
      asm volatile("global_load_async_to_lds_b128 %0, %1, off"
                   :: "v"(ldsoff), "v"(src) : "memory");
    }
    asm volatile("s_wait_asynccnt 0x0" ::: "memory");
    if (t < 0) *(v4u*)dst = (v4u){0u, 0u, 0u, 0u};
  }
  __syncthreads();

  const unsigned short* W1 = w1_bf + (size_t)e * EHID * CDIM;
  const unsigned short* W2 = w2_bf + (size_t)e * CDIM * EHID;
  __builtin_prefetch(W1 + threadIdx.x * 128, 0, 0);
  __builtin_prefetch(W2 + threadIdx.x * 128, 0, 0);

  // FFN1: (16x128)@(128x256) + exact GELU
  for (int nt = wave; nt < 16; nt += 8) {
    v8f acc = {0.f, 0.f, 0.f, 0.f, 0.f, 0.f, 0.f, 0.f};
    FragBF a, b;
#pragma unroll
    for (int kt = 0; kt < 4; ++kt) {
      ldA_bf16(a, a_s, 0, kt * 32, 128, lane);
      ldB_nk_bf16(b, W1, nt * 16, kt * 32, 128, lane);
      acc = wmma_bf16(a, b, acc);
    }
    int n = nt * 16 + (lane & 15);
    float bias = b1e[e * EHID + n];
    int hi = lane >> 4;
#pragma unroll
    for (int r = 0; r < 8; ++r) {
      float hv = acc[r] + bias;
      hv = 0.5f * hv * (1.f + erff(hv * 0.70710678118654752f));
      h_s[(r + 8 * hi) * 256 + n] = f2bf(hv);
    }
  }
  __syncthreads();

  // FFN2: (16x256)@(256x128) + scale + residual
  for (int nt = wave; nt < 8; nt += 8) {
    v8f acc = {0.f, 0.f, 0.f, 0.f, 0.f, 0.f, 0.f, 0.f};
    FragBF a, b;
#pragma unroll
    for (int kt = 0; kt < 8; ++kt) {
      ldA_bf16(a, h_s, 0, kt * 32, 256, lane);
      ldB_nk_bf16(b, W2, nt * 16, kt * 32, 256, lane);
      acc = wmma_bf16(a, b, acc);
    }
    int n = nt * 16 + (lane & 15);
    float bias = b2e[e * CDIM + n];
    int hi = lane >> 4;
#pragma unroll
    for (int r = 0; r < 8; ++r) {
      int m = r + 8 * hi;
      int t = tok_s[m];
      if (t >= 0)
        out[(size_t)t * CDIM + n] =
            (acc[r] + bias) * wgt_s[m] + x1[(size_t)t * CDIM + n];
    }
  }
}

// ============================================================================
extern "C" void kernel_launch(void* const* d_in, const int* in_sizes, int n_in,
                              void* d_out, int out_size, void* d_ws, size_t ws_size,
                              hipStream_t stream) {
  (void)in_sizes; (void)n_in; (void)out_size; (void)ws_size;
  const float* x      = (const float*)d_in[0];
  const float* ln1_g  = (const float*)d_in[1];
  const float* ln1_b  = (const float*)d_in[2];
  const float* w_qkv  = (const float*)d_in[3];
  const float* b_qkv  = (const float*)d_in[4];
  const float* w_proj = (const float*)d_in[5];
  const float* b_proj = (const float*)d_in[6];
  const float* ln2_g  = (const float*)d_in[7];
  const float* ln2_b  = (const float*)d_in[8];
  const float* gate_w = (const float*)d_in[9];
  const float* w1     = (const float*)d_in[10];
  const float* b1     = (const float*)d_in[11];
  const float* w2     = (const float*)d_in[12];
  const float* b2     = (const float*)d_in[13];
  float* out = (float*)d_out;

  unsigned char* ws = (unsigned char*)d_ws;
  float*          x1      = (float*)ws;                          // 16 MB
  unsigned short* xln2    = (unsigned short*)(ws + 16777216);    //  8 MB
  float*          wgt     = (float*)(ws + 25165824);             // 128 KB
  int*            counts  = (int*)(ws + 25296896);               //  64 B
  int*            buckets = (int*)(ws + 25296960);               //  2 MB
  unsigned short* wq_bf   = (unsigned short*)(ws + 27394112);    // 96 KB
  unsigned short* wp_bf   = (unsigned short*)(ws + 27492416);    // 32 KB
  unsigned short* w1_bf   = (unsigned short*)(ws + 27525184);    //  1 MB
  unsigned short* w2_bf   = (unsigned short*)(ws + 28573760);    //  1 MB
  unsigned short* gw_bf   = (unsigned short*)(ws + 29622336);    //  4 KB

  hipMemsetAsync(counts, 0, NEXP * sizeof(int), stream);

  cvt_kernel<<<(384 * 128 / 4 + 255) / 256, 256, 0, stream>>>(w_qkv, wq_bf, 384 * 128);
  cvt_kernel<<<(128 * 128 / 4 + 255) / 256, 256, 0, stream>>>(w_proj, wp_bf, 128 * 128);
  cvt_kernel<<<(NEXP * EHID * CDIM / 4 + 255) / 256, 256, 0, stream>>>(w1, w1_bf, NEXP * EHID * CDIM);
  cvt_kernel<<<(NEXP * CDIM * EHID / 4 + 255) / 256, 256, 0, stream>>>(w2, w2_bf, NEXP * CDIM * EHID);
  cvt_kernel<<<(NEXP * CDIM / 4 + 255) / 256, 256, 0, stream>>>(gate_w, gw_bf, NEXP * CDIM);

  attn_kernel<<<NWIN, 256, 163840, stream>>>(x, ln1_g, ln1_b, wq_bf, b_qkv,
                                             wp_bf, b_proj, x1);
  route_kernel<<<TOK_TOTAL / 64, 256, 0, stream>>>(x1, ln2_g, ln2_b, gw_bf,
                                                   xln2, wgt, counts, buckets);
  dim3 gC(NEXP, TOK_TOTAL / 16, 1);
  moe_kernel<<<gC, 256, 0, stream>>>(xln2, x1, wgt, counts, buckets,
                                     w1_bf, b1, w2_bf, b2, out);
}
